// GRU_17841294147987
// MI455X (gfx1250) — compile-verified, matrix operand order checked by reading, not run on hardware
//
#include <hip/hip_runtime.h>
#include <cmath>

// ---------------------------------------------------------------------------
// Bidirectional 2-layer GRU for MI455X (gfx1250, wave32).
//   * bf16 WMMA (v_wmma_f32_16x16x32_bf16) with f32 accumulation everywhere.
//   * Input projections: time-parallel GEMM; the per-block shared 64x32 weight
//     tile is staged into LDS with GLOBAL_LOAD_ASYNC_TO_LDS_B128 (ASYNCcnt,
//     s_wait_asynccnt), double-buffered so staging overlaps WMMA.
//   * Recurrence: h kept bf16 in LDS (double buffered) for WMMA A-fragments
//     and f32 in registers for the gate blend (producer lane == consumer lane
//     across steps -> zero global traffic for the f32 hidden state).
// Input flattening assumption (setup_inputs insertion order):
//   d_in[0]=x, d_in[1]=mask, then per cell (L0.fw, L0.bw, L1.fw, L1.bw):
//   Wxzr, bzr, Whzr, Wxh, bh, Whh.
// ---------------------------------------------------------------------------

typedef __bf16 bf16;
typedef __attribute__((ext_vector_type(16))) __bf16 v16bf;
typedef __attribute__((ext_vector_type(8)))  __bf16 bf16x8;
typedef __attribute__((ext_vector_type(8)))  float  v8f;

union FragU { v16bf v; bf16x8 h[2]; };

static __device__ __forceinline__ v16bf make_frag(const bf16* p0, const bf16* p1) {
    FragU f;
    f.h[0] = *(const bf16x8*)p0;   // 16B: K = k0+[0,8)  (lanes 0-15) / k0+[8,16)  (lanes 16-31)
    f.h[1] = *(const bf16x8*)p1;   // 16B: K = k0+[16,24)             / k0+[24,32)
    return f.v;
}

static __device__ __forceinline__ v8f wmma_bf16(v16bf a, v16bf b, v8f c) {
    // D = A(16x32) * B(32x16) + C(16x16 f32)
    return __builtin_amdgcn_wmma_f32_16x16x32_bf16(false, a, false, b, (short)0, c, false, false);
}

// Async copy of one 16B chunk global -> LDS (CDNA5 TDM-class async path).
// VDST operand is the wave-relative LDS byte offset; LDS aperture addresses
// carry that offset in the low 32 bits of the generic pointer.
static __device__ __forceinline__ void async_g2l_b128(const bf16* gsrc, bf16* ldst) {
    asm volatile("global_load_async_to_lds_b128 %0, %1, off"
                 :: "v"((unsigned int)(unsigned long long)ldst), "v"(gsrc)
                 : "memory");
}
static __device__ __forceinline__ void wait_async0() {
    asm volatile("s_wait_asynccnt 0x0" ::: "memory");
}

// --------------------------- helper kernels --------------------------------

__global__ void cvt_f32_bf16(const float* __restrict__ s, bf16* __restrict__ d, int n) {
    for (int i = blockIdx.x * blockDim.x + threadIdx.x; i < n; i += gridDim.x * blockDim.x)
        d[i] = (bf16)s[i];
}

__global__ void copy_f32(const float* __restrict__ s, float* __restrict__ d, int n) {
    for (int i = blockIdx.x * blockDim.x + threadIdx.x; i < n; i += gridDim.x * blockDim.x)
        d[i] = s[i];
}

// ------------------- phase 1: input projection GEMM ------------------------
// gates(M x 1536) = Xbf(M x K) @ W(1536 x K)^T + bias.  Block = 8 waves.
// Wave w owns rows [blockM*128 + w*16, +16); all waves share one 64-col N
// group whose 64x32 bf16 weight tile is async-staged into LDS per k-step.

__global__ void __launch_bounds__(256)
gru_xproj_wmma(const bf16* __restrict__ X, int K,
               const bf16* __restrict__ W, const float* __restrict__ bias,
               float* __restrict__ gates)
{
    const int G = 1536;
    const int PAD = 40;                                  // padded 32-k row, dodges bank conflicts
    __shared__ __align__(16) bf16 btile[2][64 * PAD];    // double-buffered B tile (10 KB)

    const int tid  = threadIdx.x;
    const int lane = tid & 31;
    const int wave = tid >> 5;
    const int l15  = lane & 15;
    const int hi   = lane >> 4;                          // 0: lanes 0-15, 1: lanes 16-31

    const int m0 = blockIdx.x * 128 + wave * 16;
    const int n0 = blockIdx.y * 64;

    // staging assignment: each thread moves one 16B chunk of the 64x32 tile
    const int srow = tid >> 2;                           // W row within tile (0..63)
    const int schk = tid & 3;                            // 16B chunk within row (0..3)
    const bf16* sgp = W + (size_t)(n0 + srow) * K + schk * 8;
    bf16* sl[2] = { &btile[0][srow * PAD + schk * 8],
                    &btile[1][srow * PAD + schk * 8] };

    v8f acc[4] = {};
    const bf16* arow = X + (size_t)(m0 + l15) * K + hi * 8;

    // prologue: stage k=0 tile into buffer 0
    async_g2l_b128(sgp, sl[0]);

    for (int k0 = 0; k0 < K; k0 += 32) {
        const int cur = (k0 >> 5) & 1;
        wait_async0();                                   // own async write landed in LDS
        __syncthreads();                                 // all waves' stages + prior reads done
        if (k0 + 32 < K)
            async_g2l_b128(sgp + k0 + 32, sl[cur ^ 1]);  // overlap next stage with WMMA

        v16bf a = make_frag(arow + k0, arow + k0 + 16);
        const bf16* bbase = &btile[cur][0];
#pragma unroll
        for (int nt = 0; nt < 4; ++nt) {
            const bf16* bl = bbase + (nt * 16 + l15) * PAD + hi * 8;
            v16bf b = make_frag(bl, bl + 16);
            acc[nt] = wmma_bf16(a, b, acc[nt]);
        }
    }

#pragma unroll
    for (int nt = 0; nt < 4; ++nt) {
        const int n  = n0 + nt * 16 + l15;
        const float bn = bias[n];
#pragma unroll
        for (int r = 0; r < 8; ++r) {
            const int m = m0 + r + hi * 8;               // C/D layout: vgpr r -> M=r (+8 for hi lanes)
            gates[(size_t)m * G + n] = acc[nt][r] + bn;
        }
    }
}

// ----------------------- phase 2: GRU recurrence ---------------------------
// Grid: (4 batch tiles, 2 directions). Block: 32 waves. Each block owns 16
// batch rows; wave w owns hidden columns [w*16, w*16+16) and its 3 gate tiles.

__global__ void __launch_bounds__(1024)
gru_recur_wmma(const float* __restrict__ gates_fw, const float* __restrict__ gates_bw,
               const bf16* __restrict__ Wh_fw, const bf16* __restrict__ Wh_bw,
               const float* __restrict__ mask,
               bf16* __restrict__ out_bf, float* __restrict__ out_f32,
               int T, int B)
{
    const int H = 512, G = 1536, HP = H + 8;             // padded rows vs bank conflicts
    __shared__ __align__(16) bf16 hbuf[2][16][520];

    const int lane = threadIdx.x & 31;
    const int wave = threadIdx.x >> 5;
    const int l15  = lane & 15;
    const int hi   = lane >> 4;

    const int m0   = blockIdx.x * 16;                    // batch tile base
    const int dir  = blockIdx.y;                         // 0 = forward, 1 = backward
    const float* gates = dir ? gates_bw : gates_fw;
    const bf16*  Wh    = dir ? Wh_bw : Wh_fw;            // rows 0..1023: Whzr, 1024..1535: Whh
    const int ocol = dir * H;
    const int j0   = wave * 16;                          // 32 waves * 16 = H
    const int j    = j0 + l15;

    {   // h0 = 0
        bf16* hz = &hbuf[0][0][0];
        for (int i = threadIdx.x; i < 16 * HP; i += blockDim.x) hz[i] = (bf16)0.0f;
    }
    float hprev[8];
#pragma unroll
    for (int r = 0; r < 8; ++r) hprev[r] = 0.0f;

    const bf16* wz = Wh + (size_t)(j0 + l15) * H + hi * 8;          // z rows
    const bf16* wr = Wh + (size_t)(512 + j0 + l15) * H + hi * 8;    // r rows
    const bf16* wc = Wh + (size_t)(1024 + j0 + l15) * H + hi * 8;   // candidate rows

    __syncthreads();

    for (int s = 0; s < T; ++s) {
        const int t   = dir ? (T - 1 - s) : s;
        const int cur = s & 1, nxt = cur ^ 1;

        // warm the gate row for this step while WMMAs run
        __builtin_prefetch(gates + ((size_t)t * B + m0 + (lane & 7)) * G + j0, 0, 1);

        v8f az = {}, ar = {}, ac = {};
        const bf16* hrow = &hbuf[cur][l15][0] + hi * 8;
        for (int k0 = 0; k0 < H; k0 += 32) {
            v16bf a = make_frag(hrow + k0, hrow + k0 + 16);
            az = wmma_bf16(a, make_frag(wz + k0, wz + k0 + 16), az);
            ar = wmma_bf16(a, make_frag(wr + k0, wr + k0 + 16), ar);
            ac = wmma_bf16(a, make_frag(wc + k0, wc + k0 + 16), ac);
        }

        // z/r/h~ tiles share the C/D lane layout -> purely per-lane update.
        // Writes go to the *other* LDS buffer: no barrier before them needed.
#pragma unroll
        for (int r = 0; r < 8; ++r) {
            const int b = m0 + r + hi * 8;
            const size_t row = (size_t)t * B + b;
            const float gz = gates[row * G + j];
            const float gr = gates[row * G + 512 + j];
            const float gc = gates[row * G + 1024 + j];
            const float z  = 1.0f / (1.0f + __expf(-(az[r] + gz)));
            const float rr = 1.0f / (1.0f + __expf(-(ar[r] + gr)));
            const float hc = tanhf(gc + rr * ac[r]);     // r gates h@Whh^T before tanh
            float hn = (1.0f - z) * hc + z * hprev[r];
            const float mt = mask[row];
            hn = mt * hn + (1.0f - mt) * hprev[r];
            hprev[r] = hn;
            hbuf[nxt][r + hi * 8][j] = (bf16)hn;
            if (out_bf)  out_bf [row * (size_t)(2 * H) + ocol + j] = (bf16)hn;
            if (out_f32) out_f32[row * (size_t)(2 * H) + ocol + j] = hn;
        }
        __syncthreads();
    }
}

// ------------------------------- launcher ----------------------------------

extern "C" void kernel_launch(void* const* d_in, const int* in_sizes, int n_in,
                              void* d_out, int out_size, void* d_ws, size_t ws_size,
                              hipStream_t stream)
{
    const int T = 256, Bb = 64, H = 512, G = 1536;
    const size_t M = (size_t)T * Bb;                     // 16384 rows

    // workspace carve-out (~254 MB)
    char* ws = (char*)d_ws;
    size_t off = 0;
    auto take = [&](size_t bytes) -> char* {
        char* p = ws + off;
        off = (off + bytes + 255) & ~(size_t)255;
        return p;
    };
    bf16*  XBF = (bf16*)take(M * 1024 * 2);              // layer input; layer0 output lands here
    bf16*  WX[4]; for (int c = 0; c < 4; ++c) WX[c] = (bf16*) take((size_t)G * 1024 * 2);
    bf16*  WH[4]; for (int c = 0; c < 4; ++c) WH[c] = (bf16*) take((size_t)G * H * 2);
    float* BI[4]; for (int c = 0; c < 4; ++c) BI[c] = (float*)take((size_t)G * 4);
    float* GFW = (float*)take(M * G * 4);
    float* GBW = (float*)take(M * G * 4);

    const float* x    = (const float*)d_in[0];
    const float* mask = (const float*)d_in[1];

    auto cvt = [&](const float* s, bf16* d, size_t n) {
        int bl = 256;
        int gr = (int)((n + bl - 1) / bl); if (gr > 4096) gr = 4096;
        cvt_f32_bf16<<<dim3(gr), dim3(bl), 0, stream>>>(s, d, (int)n);
    };
    auto cpy = [&](const float* s, float* d, size_t n) {
        copy_f32<<<dim3(8), dim3(256), 0, stream>>>(s, d, (int)n);
    };

    // stage bf16 activations + fused weight layouts
    cvt(x, XBF, M * 512);
    const int Din[2] = {512, 1024};
    for (int L = 0; L < 2; ++L) {
        for (int Dd = 0; Dd < 2; ++Dd) {
            const int c = L * 2 + Dd;
            const int base = 2 + c * 6;                  // Wxzr, bzr, Whzr, Wxh, bh, Whh
            const int K = Din[L];
            cvt((const float*)d_in[base + 0], WX[c],                    (size_t)1024 * K);
            cvt((const float*)d_in[base + 3], WX[c] + (size_t)1024 * K, (size_t)512  * K);
            cvt((const float*)d_in[base + 2], WH[c],                    (size_t)1024 * H);
            cvt((const float*)d_in[base + 5], WH[c] + (size_t)1024 * H, (size_t)512  * H);
            cpy((const float*)d_in[base + 1], BI[c],        1024);
            cpy((const float*)d_in[base + 4], BI[c] + 1024,  512);
        }
    }

    const dim3 ggrid(128, 24), gblk(256);                // M/128 x G/64
    const dim3 rgrid(4, 2),   rblk(1024);                // batch tiles x directions

    // ---- layer 0 (Din = 512) ----
    gru_xproj_wmma<<<ggrid, gblk, 0, stream>>>(XBF, 512, WX[0], BI[0], GFW);
    gru_xproj_wmma<<<ggrid, gblk, 0, stream>>>(XBF, 512, WX[1], BI[1], GBW);
    // recurrence writes bf16 concat(out_fw, out_bw) straight into XBF (layer1 input)
    gru_recur_wmma<<<rgrid, rblk, 0, stream>>>(GFW, GBW, WH[0], WH[1], mask,
                                               XBF, (float*)nullptr, T, Bb);

    // ---- layer 1 (Din = 1024) ----
    gru_xproj_wmma<<<ggrid, gblk, 0, stream>>>(XBF, 1024, WX[2], BI[2], GFW);
    gru_xproj_wmma<<<ggrid, gblk, 0, stream>>>(XBF, 1024, WX[3], BI[3], GBW);
    gru_recur_wmma<<<rgrid, rblk, 0, stream>>>(GFW, GBW, WH[2], WH[3], mask,
                                               (bf16*)nullptr, (float*)d_out, T, Bb);

    (void)in_sizes; (void)n_in; (void)out_size; (void)ws_size;
}
